// LSTMModel_24550033063990
// MI455X (gfx1250) — compile-verified, matrix operand order checked by reading, not run on hardware
//
#include <hip/hip_runtime.h>
#include <hip/hip_bf16.h>

// LSTM: T=1024, B=64, IN=128, H=1024, OUT=128.
// Persistent-RNN design for MI455X (gfx1250, wave32, WMMA bf16 16x16x32):
//  - 64 persistent WGs x 128 threads (4 waves). WG w owns hidden columns
//    j0=w*16 for all four gates; wave v owns batch rows v*16..v*16+15.
//  - Each WG stages its W_hh/W_ih slice (64 rows x 1152 K, bf16) into LDS once;
//    all 1024 steps read weights from LDS only (padded rows -> conflict-free b128).
//  - Cell state c is held in registers (WMMA C-fragment) for the whole sequence.
//  - h double-buffered in global bf16; one device-wide software barrier per step.
//  - Output projection out[t] = h_{t+1} @ W_out^T is computed by WGs 0..31
//    (wave 0) in the *next* iteration under the same barrier -> no hs buffer.

typedef __bf16 bf16_t;
typedef __attribute__((ext_vector_type(16))) __bf16 v16bf;
typedef __attribute__((ext_vector_type(8)))  float  v8f;

#define T_STEPS 1024
#define BATCH   64
#define IN_DIM  128
#define HID     1024
#define OUT_DIM 128
#define NWG     64
#define LDSK    (HID + 8)     // padded LDS row stride (elems) for W_hh slice
#define LDSKX   (IN_DIM + 8)  // padded LDS row stride (elems) for W_ih slice
#define SMEM_BYTES ((64 * LDSK + 64 * LDSKX) * 2)

union FragU { v16bf v; uint4 u[2]; };

// Load one 16x32 bf16 WMMA operand tile. Works for both A (rows of h/x) and
// B (rows of W == columns of W^T): lane l holds matrix row/col (row0 + l%16);
// lanes 0-15 carry K=k0+0..7 (v0-3) and k0+16..23 (v4-7), lanes 16-31 carry
// K=k0+8..15 and k0+24..31  -> two contiguous 16B loads per lane.
__device__ __forceinline__ v16bf load_frag(const bf16_t* __restrict__ src,
                                           int ld, int row0, int k0, int lane) {
  const int r = lane & 15, hs = lane >> 4;
  const bf16_t* p = src + (size_t)(row0 + r) * (size_t)ld + (size_t)(k0 + hs * 8);
  FragU f;
  f.u[0] = *reinterpret_cast<const uint4*>(p);
  f.u[1] = *reinterpret_cast<const uint4*>(p + 16);
  return f.v;
}

__device__ __forceinline__ v8f wmma_bf16(v16bf a, v16bf b, v8f c) {
  return __builtin_amdgcn_wmma_f32_16x16x32_bf16(false, a, false, b, (short)0, c,
                                                 false, false);
}

__device__ __forceinline__ float sigmoid_f(float x) {
  return 1.0f / (1.0f + __expf(-x));
}
__device__ __forceinline__ float tanh_f(float x) {
  return 1.0f - 2.0f / (__expf(2.0f * x) + 1.0f);
}

// Sense-reversing device-wide barrier. bar[0]=arrival count, bar[1]=generation.
__device__ __forceinline__ void grid_barrier(unsigned* bar, unsigned nwg) {
  __threadfence();          // make this thread's global stores device-visible
  __syncthreads();
  if (threadIdx.x == 0) {
    volatile unsigned* gen = bar + 1;
    unsigned g = *gen;
    if (atomicAdd(bar, 1u) == nwg - 1u) {
      bar[0] = 0u;
      __threadfence();
      atomicAdd((unsigned*)gen, 1u);
    } else {
      while (*gen == g) { __builtin_amdgcn_s_sleep(2); }
    }
    __threadfence();
  }
  __syncthreads();
}

__global__ void f32_to_bf16_kernel(const float* __restrict__ src,
                                   bf16_t* __restrict__ dst, int n) {
  int i = blockIdx.x * blockDim.x + threadIdx.x;
  if (i < n) dst[i] = (bf16_t)src[i];
}

__global__ void init_barrier_kernel(unsigned* bar) {
  if (threadIdx.x < 2) bar[threadIdx.x] = 0u;
}

__global__ void __launch_bounds__(128, 1)
lstm_persistent_kernel(const bf16_t* __restrict__ in_bf,   // (T,B,IN) bf16
                       const float*  __restrict__ c0,      // (B,H) f32
                       const bf16_t* __restrict__ Whh_bf,  // (4H,H) bf16
                       const bf16_t* __restrict__ Wih_bf,  // (4H,IN) bf16
                       const float*  __restrict__ b_ih,    // (4H)
                       const float*  __restrict__ b_hh,    // (4H)
                       const bf16_t* __restrict__ Wout_bf, // (OUT,H) bf16
                       const float*  __restrict__ b_out,   // (OUT)
                       bf16_t* __restrict__ hbuf,          // 2 x (B,H) bf16
                       float*  __restrict__ out,           // (T,B,OUT) f32
                       unsigned* __restrict__ bar) {
  extern __shared__ __align__(16) unsigned char smem_raw[];
  bf16_t* Whh_s = reinterpret_cast<bf16_t*>(smem_raw);  // 64 rows x LDSK
  bf16_t* Wih_s = Whh_s + 64 * LDSK;                    // 64 rows x LDSKX

  const int wg   = blockIdx.x;    // 0..63 : hidden-column tile
  const int tid  = threadIdx.x;   // 0..127
  const int lane = tid & 31;
  const int wave = tid >> 5;      // 0..3 : batch tile
  const int hsel = lane >> 4;     // C/D: row m = r + 8*hsel
  const int n    = lane & 15;     // C/D: column
  const int j0   = wg * 16;
  const int m0   = wave * 16;

  // ---- stage this WG's weight slice into LDS (reused for all T steps) ----
  // LDS row (g*16 + jj) = global W row (g*HID + j0 + jj).
  for (int idx = tid; idx < 64 * (HID / 8); idx += 128) {
    int row = idx / (HID / 8);
    int c8  = idx % (HID / 8);
    int grow = (row >> 4) * HID + j0 + (row & 15);
    *reinterpret_cast<uint4*>(Whh_s + row * LDSK + c8 * 8) =
        *reinterpret_cast<const uint4*>(Whh_bf + (size_t)grow * HID + c8 * 8);
  }
  for (int idx = tid; idx < 64 * (IN_DIM / 8); idx += 128) {
    int row = idx / (IN_DIM / 8);
    int c8  = idx % (IN_DIM / 8);
    int grow = (row >> 4) * HID + j0 + (row & 15);
    *reinterpret_cast<uint4*>(Wih_s + row * LDSKX + c8 * 8) =
        *reinterpret_cast<const uint4*>(Wih_bf + (size_t)grow * IN_DIM + c8 * 8);
  }
  __syncthreads();

  // Per-lane gate biases (depend only on column n).
  float bias[4];
#pragma unroll
  for (int g = 0; g < 4; ++g) {
    int col = g * HID + j0 + n;
    bias[g] = b_ih[col] + b_hh[col];
  }

  // Cell state fragment, persistent in registers for the whole sequence.
  v8f c = {0.f, 0.f, 0.f, 0.f, 0.f, 0.f, 0.f, 0.f};
#pragma unroll
  for (int r = 0; r < 8; ++r)
    c[r] = c0[(size_t)(m0 + r + 8 * hsel) * HID + (j0 + n)];

  // Output-projection tile owned by WGs 0..31 (wave 0): out is (B=64, OUT=128).
  const int otm = (wg >> 3) * 16;  // batch-row tile
  const int otn = (wg & 7) * 16;   // out-col tile
  const float ob = (wg < 32) ? b_out[otn + n] : 0.0f;

  for (int t = 0; t < T_STEPS; ++t) {
    const bf16_t* hcur = hbuf + (size_t)(t & 1) * (BATCH * HID);
    bf16_t*       hnxt = hbuf + (size_t)((t & 1) ^ 1) * (BATCH * HID);

    // out[t-1] = h_t @ W_out^T + b_out  (h_t resident in hcur)
    if (t > 0 && wg < 32 && wave == 0) {
      v8f acc = {0.f, 0.f, 0.f, 0.f, 0.f, 0.f, 0.f, 0.f};
      for (int kb = 0; kb < HID / 32; ++kb) {
        v16bf a = load_frag(hcur, HID, otm, kb * 32, lane);
        v16bf b = load_frag(Wout_bf, HID, otn, kb * 32, lane);
        acc = wmma_bf16(a, b, acc);
      }
      float* orow = out + (size_t)(t - 1) * (BATCH * OUT_DIM);
#pragma unroll
      for (int r = 0; r < 8; ++r)
        orow[(size_t)(otm + r + 8 * hsel) * OUT_DIM + (otn + n)] = acc[r] + ob;
    }

    // gates = [h | x_t] @ [W_hh | W_ih]^T  for this WG's 16 columns x 4 gates
    v8f acc0 = {0.f, 0.f, 0.f, 0.f, 0.f, 0.f, 0.f, 0.f};
    v8f acc1 = acc0, acc2 = acc0, acc3 = acc0;
    for (int kb = 0; kb < HID / 32; ++kb) {
      v16bf a = load_frag(hcur, HID, m0, kb * 32, lane);
      acc0 = wmma_bf16(a, load_frag(Whh_s, LDSK, 0, kb * 32, lane), acc0);
      acc1 = wmma_bf16(a, load_frag(Whh_s, LDSK, 16, kb * 32, lane), acc1);
      acc2 = wmma_bf16(a, load_frag(Whh_s, LDSK, 32, kb * 32, lane), acc2);
      acc3 = wmma_bf16(a, load_frag(Whh_s, LDSK, 48, kb * 32, lane), acc3);
    }
    const bf16_t* xt = in_bf + (size_t)t * (BATCH * IN_DIM);
    for (int kb = 0; kb < IN_DIM / 32; ++kb) {
      v16bf a = load_frag(xt, IN_DIM, m0, kb * 32, lane);
      acc0 = wmma_bf16(a, load_frag(Wih_s, LDSKX, 0, kb * 32, lane), acc0);
      acc1 = wmma_bf16(a, load_frag(Wih_s, LDSKX, 16, kb * 32, lane), acc1);
      acc2 = wmma_bf16(a, load_frag(Wih_s, LDSKX, 32, kb * 32, lane), acc2);
      acc3 = wmma_bf16(a, load_frag(Wih_s, LDSKX, 48, kb * 32, lane), acc3);
    }

    // Pointwise LSTM cell update (i,f,g,o gate order), c kept in registers.
#pragma unroll
    for (int r = 0; r < 8; ++r) {
      float ig = sigmoid_f(acc0[r] + bias[0]);
      float fg = sigmoid_f(acc1[r] + bias[1]);
      float gg = tanh_f(acc2[r] + bias[2]);
      float og = sigmoid_f(acc3[r] + bias[3]);
      float cv = fg * c[r] + ig * gg;
      c[r] = cv;
      float hv = og * tanh_f(cv);
      hnxt[(size_t)(m0 + r + 8 * hsel) * HID + (j0 + n)] = (bf16_t)hv;
    }

    grid_barrier(bar, NWG);
  }

  // Final projection: out[T-1] from hbuf[0] (written at t = T-1).
  if (wg < 32 && wave == 0) {
    const bf16_t* hfin = hbuf;  // ((T-1)&1)^1 == 0
    v8f acc = {0.f, 0.f, 0.f, 0.f, 0.f, 0.f, 0.f, 0.f};
    for (int kb = 0; kb < HID / 32; ++kb) {
      v16bf a = load_frag(hfin, HID, otm, kb * 32, lane);
      v16bf b = load_frag(Wout_bf, HID, otn, kb * 32, lane);
      acc = wmma_bf16(a, b, acc);
    }
    float* orow = out + (size_t)(T_STEPS - 1) * (BATCH * OUT_DIM);
#pragma unroll
    for (int r = 0; r < 8; ++r)
      orow[(size_t)(otm + r + 8 * hsel) * OUT_DIM + (otn + n)] = acc[r] + ob;
  }
}

extern "C" void kernel_launch(void* const* d_in, const int* in_sizes, int n_in,
                              void* d_out, int out_size, void* d_ws, size_t ws_size,
                              hipStream_t stream) {
  const float* in_seq = (const float*)d_in[0];  // (T,B,IN)
  const float* h0     = (const float*)d_in[1];  // (B,H)
  const float* c0     = (const float*)d_in[2];  // (B,H)
  const float* W_ih   = (const float*)d_in[3];  // (4H,IN)
  const float* W_hh   = (const float*)d_in[4];  // (4H,H)
  const float* b_ih   = (const float*)d_in[5];
  const float* b_hh   = (const float*)d_in[6];
  const float* W_out  = (const float*)d_in[7];  // (OUT,H)
  const float* b_out  = (const float*)d_in[8];
  float* out = (float*)d_out;

  // Workspace layout (~26.7 MB total; 256B-aligned slices).
  char* ws = (char*)d_ws;
  size_t off = 0;
  auto carve = [&](size_t bytes) {
    void* p = ws + off;
    off += (bytes + 255) & ~(size_t)255;
    return p;
  };
  unsigned* bar     = (unsigned*)carve(2 * sizeof(unsigned));
  bf16_t*   Whh_bf  = (bf16_t*)carve((size_t)4 * HID * HID * 2);
  bf16_t*   Wih_bf  = (bf16_t*)carve((size_t)4 * HID * IN_DIM * 2);
  bf16_t*   Wout_bf = (bf16_t*)carve((size_t)OUT_DIM * HID * 2);
  bf16_t*   in_bf   = (bf16_t*)carve((size_t)T_STEPS * BATCH * IN_DIM * 2);
  bf16_t*   hbuf    = (bf16_t*)carve((size_t)2 * BATCH * HID * 2);
  (void)ws_size; (void)in_sizes; (void)n_in; (void)out_size;

  auto cvt = [&](const float* s, bf16_t* d, int n) {
    f32_to_bf16_kernel<<<(n + 255) / 256, 256, 0, stream>>>(s, d, n);
  };
  cvt(W_hh,  Whh_bf,  4 * HID * HID);
  cvt(W_ih,  Wih_bf,  4 * HID * IN_DIM);
  cvt(W_out, Wout_bf, OUT_DIM * HID);
  cvt(in_seq, in_bf,  T_STEPS * BATCH * IN_DIM);
  cvt(h0,    hbuf,    BATCH * HID);  // hbuf[0] = h0 ; hbuf[1] written at t=0
  init_barrier_kernel<<<1, 32, 0, stream>>>(bar);

  lstm_persistent_kernel<<<NWG, 128, SMEM_BYTES, stream>>>(
      in_bf, c0, Whh_bf, Wih_bf, b_ih, b_hh, Wout_bf, b_out, hbuf, out, bar);
}
